// VectorQuantizer_67954972557401
// MI455X (gfx1250) — matrix operand order repaired and checked by reading, not compiled, not run on hardware
//
#include <hip/hip_runtime.h>
#include <math.h>

// ---------------------------------------------------------------------------
// VQ-VAE vector quantizer for MI455X (gfx1250, wave32, WMMA).
//
// Shapes: inputs [16, 64, 8192] f32, codebook [512, 64] f32.
// N = 131072 vectors of dim 64, K = 512 entries.
//
// Roofline: distance matmul is only ~8.6 GFLOP; traffic (~370MB, dominated by
// the 268MB one-hot matrix) sets a ~16us floor at 23.3 TB/s. WMMA
// (v_wmma_f32_16x16x32_f16) makes the matmul negligible vs that floor.
// Argmin needs only (||e||^2 - 2 x.e); f16 inputs + f32 accumulation keep the
// argmin stable (score gaps ~1e-2 vs conversion error ~1e-5).
//
// The 64KB f16-swizzled codebook is staged LDS-resident per WG (320KB LDS per
// WGP on CDNA5) via GLOBAL_LOAD_ASYNC_TO_LDS_B128 (ASYNCcnt-tracked), so the
// inner loop reads A-operands with 4 conflict-free ds_load_b128 instead of
// 16 global b32 loads, keeping the VMEM path free for the streaming stores.
// ---------------------------------------------------------------------------

typedef __attribute__((ext_vector_type(16))) _Float16 v16h;
typedef __attribute__((ext_vector_type(8)))  float    v8f;
typedef __attribute__((ext_vector_type(4)))  int      vq_i4;

typedef vq_i4 __attribute__((address_space(1)))* vq_gp_i4;  // global <4 x i32>*
typedef vq_i4 __attribute__((address_space(3)))* vq_lp_i4;  // LDS    <4 x i32>*

#define NUM_EMB 512
#define EMB_DIM 64
#define BB      16
#define WW      8192
#define NVEC    (BB * WW)            /* 131072 */
#define QELEMS  (BB * EMB_DIM * WW)  /* 8388608 */

// d_out layout (floats), reference return order:
// embedding_loss, commitment_loss, quantized_out, perplexity, encodings, indices
#define O_EL  0
#define O_CL  1
#define O_Q   2
#define O_P   ((size_t)2 + QELEMS)                       /* 8388610 */
#define O_ENC (O_P + 1)                                  /* 8388611 */
#define O_IDX (O_ENC + (size_t)NVEC * NUM_EMB)           /* 75497475 */

// workspace layout (bytes)
#define WS_A   0        /* swizzled f16 codebook A-operand: 16384 dwords = 64KB */
#define WS_E2  65536    /* 512 floats: ||e||^2 */
#define WS_CNT 67584    /* 512 u32: histogram */
#define WS_LP  69632    /* 1024 floats: per-WG loss partials */

// CDNA5 async global->LDS copy path (guarded; falls back to load+store).
#if defined(__has_builtin)
#if __has_builtin(__builtin_amdgcn_global_load_async_to_lds_b128) && \
    __has_builtin(__builtin_amdgcn_s_wait_asynccnt)
#define VQ_ASYNC_LDS 1
#endif
#endif

// ---------------------------------------------------------------------------
// K0: build the A-operand swizzle table, ||e||^2, zero histogram.
// 64 blocks x 256 = 16384 threads = one thread per table dword.
//
// Table layout (dword index): ((blk*4 + q)*32 + lane)*4 + j, kk = 4q + j.
// This makes the main kernel's per-block fetch 4 uint4 LDS loads with a
// 16B lane stride (conflict-free b128 pattern).
//
// WMMA 16-bit A layout (ISA 7.12.2), lane L (half=L/16, M=L%16), VGPR v holds
// packed {K0, K0+1}:  v<4: K0 = 2v + 8*half ;  v>=4: K0 = 16 + 2(v-4) + 8*half.
// Two chained WMMAs cover K = 0..31 (w=0) and 32..63 (w=1); kk = 8w + v.
// ---------------------------------------------------------------------------
__global__ void __launch_bounds__(256) vq_init(const float* __restrict__ cb,
                                               unsigned* __restrict__ wsA,
                                               float* __restrict__ e2,
                                               unsigned* __restrict__ cnt) {
    int gid  = blockIdx.x * 256 + threadIdx.x;
    int j    = gid & 3;
    int lane = (gid >> 2) & 31;
    int qq   = (gid >> 7) & 3;
    int blk  = gid >> 9;               // 0..31 codebook block of 16 entries
    int kk   = 4 * qq + j;             // 0..15
    int w    = kk >> 3;                // which of the two WMMAs (K offset 32w)
    int v    = kk & 7;
    int half = lane >> 4;
    int M    = lane & 15;
    int k0   = 32 * w + ((v < 4) ? (2 * v + 8 * half)
                                 : (16 + 2 * (v - 4) + 8 * half));
    int entry = blk * 16 + M;
    _Float16 h0 = (_Float16)cb[entry * EMB_DIM + k0];
    _Float16 h1 = (_Float16)cb[entry * EMB_DIM + k0 + 1];
    unsigned u = ((unsigned)__builtin_bit_cast(unsigned short, h1) << 16) |
                 (unsigned)__builtin_bit_cast(unsigned short, h0);
    wsA[gid] = u;

    if (gid < NUM_EMB) {
        float s = 0.0f;
        for (int c = 0; c < EMB_DIM; ++c) {
            float t = cb[gid * EMB_DIM + c];
            s += t * t;
        }
        e2[gid]  = s;
        cnt[gid] = 0u;
    }
}

// ---------------------------------------------------------------------------
// K1: main kernel. 1024 WGs x 256 threads (8 waves). Each wave owns 16
// vectors; the WG owns 128 consecutive vectors (same batch b: 8192 % 128 == 0).
// B-operand: 16 x-vectors in WMMA B layout (lane L holds column N=L%16,
// VGPR v holds K = {8*(L/16)+2v, +1} for v<4, {16+8*(L/16)+2(v-4), +1} for v>=4, +32 for the 2nd WMMA).
// Loop over 32 codebook blocks: D = A x B (2 chained f16 WMMAs, f32 acc),
// score = ||e||^2 - 2*dot, per-lane running argmin over the lane's 8 M-rows,
// then shfl_xor(16) merges halves. Quantized gather + loss partial follow.
// ---------------------------------------------------------------------------
__global__ void __launch_bounds__(256) vq_main(const float* __restrict__ x,
                                               const float* __restrict__ cb,
                                               const unsigned* __restrict__ wsA,
                                               const float* __restrict__ e2,
                                               unsigned* __restrict__ cnt,
                                               float* __restrict__ lossp,
                                               float* __restrict__ out) {
    __shared__ __align__(16) unsigned ldsA[16384];   // 64KB swizzled codebook
    __shared__ float e2s[NUM_EMB];
    __shared__ int   idx_sh[128];
    __shared__ float red[256];

    const int tid = threadIdx.x;

    // Stage the 64KB A-operand table into LDS (async path on CDNA5).
#ifdef VQ_ASYNC_LDS
    {
        vq_gp_i4 gsrc = (vq_gp_i4)wsA;    // global <4 x i32>*
        vq_lp_i4 ldst = (vq_lp_i4)ldsA;   // LDS    <4 x i32>*
        #pragma unroll
        for (int i = 0; i < 16; ++i) {
            int idx = tid + i * 256;      // 16B elements
            __builtin_amdgcn_global_load_async_to_lds_b128(gsrc + idx, ldst + idx,
                                                           0, 0);
        }
        __builtin_amdgcn_s_wait_asynccnt(0);
    }
#else
    {
        const uint4* src = (const uint4*)wsA;
        uint4*       dst = (uint4*)ldsA;
        #pragma unroll
        for (int i = 0; i < 16; ++i) {
            int idx = tid + i * 256;
            dst[idx] = src[idx];
        }
    }
#endif
    for (int i = tid; i < NUM_EMB; i += 256) e2s[i] = e2[i];
    __syncthreads();

    const int wv   = tid >> 5;
    const int lane = tid & 31;
    const int half = lane >> 4;
    const int col  = lane & 15;
    const int nbase_wg = blockIdx.x * 128;
    const int n  = nbase_wg + wv * 16 + col;
    const int b  = n >> 13;        // n / 8192
    const int wi = n & 8191;
    const float* xv = x + (size_t)b * EMB_DIM * WW + wi;

    // B-operand: this lane's vector components in WMMA B layout (f16 packed).
    v16h bmat[2];
    #pragma unroll
    for (int w = 0; w < 2; ++w) {
        #pragma unroll
        for (int v = 0; v < 8; ++v) {
            int c0 = 32 * w + ((v < 4) ? (8 * half + 2 * v) : (16 + 8 * half + 2 * (v - 4)));
            bmat[w][2 * v]     = (_Float16)xv[(size_t)c0 * WW];
            bmat[w][2 * v + 1] = (_Float16)xv[(size_t)(c0 + 1) * WW];
        }
    }

    const uint4* lA = (const uint4*)ldsA;
    float bestv = 3.4e38f;
    int   besti = 0;
    for (int blk = 0; blk < 32; ++blk) {
        union { uint4 q[4]; v16h h[2]; } am;
        #pragma unroll
        for (int q = 0; q < 4; ++q)
            am.q[q] = lA[(blk * 4 + q) * 32 + lane];   // conflict-free b128
        v8f c = {0.f, 0.f, 0.f, 0.f, 0.f, 0.f, 0.f, 0.f};
        c = __builtin_amdgcn_wmma_f32_16x16x32_f16(false, am.h[0], false, bmat[0],
                                                   (short)0, c, false, false);
        c = __builtin_amdgcn_wmma_f32_16x16x32_f16(false, am.h[1], false, bmat[1],
                                                   (short)0, c, false, false);
        const int mbase = blk * 16 + 8 * half;
        #pragma unroll
        for (int r = 0; r < 8; ++r) {
            float score = e2s[mbase + r] - 2.0f * c[r];
            if (score < bestv) { bestv = score; besti = mbase + r; }
        }
    }
    // Merge the two 16-lane halves (lane n holds M 0..7, lane n+16 holds M 8..15).
    {
        float ov = __shfl_xor(bestv, 16, 32);
        int   oi = __shfl_xor(besti, 16, 32);
        if (ov < bestv || (ov == bestv && oi < besti)) { bestv = ov; besti = oi; }
    }
    if (lane < 16) {
        idx_sh[wv * 16 + col] = besti;
        atomicAdd(&cnt[besti], 1u);
        out[O_IDX + n] = (float)besti;   // reference returns indices (cast to out dtype)
    }
    __syncthreads();

    // Quantized gather (fp32 codebook), straight-through output == q in fwd,
    // plus per-WG sum of (q - x)^2. Coalesced 128-wide writes per c-row.
    const int b_wg = nbase_wg >> 13;
    const int w0   = nbase_wg & 8191;
    float sum = 0.0f;
    #pragma unroll 4
    for (int i = 0; i < 32; ++i) {
        int flat = tid + i * 256;        // 0..8191 = 64 c-rows x 128 vectors
        int cdim = flat >> 7;
        int wl   = flat & 127;
        int qi   = idx_sh[wl];
        float q  = cb[qi * EMB_DIM + cdim];
        size_t off = (size_t)b_wg * EMB_DIM * WW + (size_t)cdim * WW + (w0 + wl);
        float d = q - x[off];
        sum += d * d;
        out[O_Q + off] = q;
    }
    red[tid] = sum;
    __syncthreads();
    for (int s = 128; s > 0; s >>= 1) {
        if (tid < s) red[tid] += red[tid + s];
        __syncthreads();
    }
    if (tid == 0) lossp[blockIdx.x] = red[0];   // deterministic: no float atomics
}

// ---------------------------------------------------------------------------
// K2: one-hot encodings writer, 268 MB streaming store (the bandwidth limiter).
// 65536 WGs x 256 threads, 4 consecutive floats each.
// ---------------------------------------------------------------------------
__global__ void __launch_bounds__(256) vq_enc(float* __restrict__ out) {
    size_t gid  = (size_t)blockIdx.x * 256 + threadIdx.x;
    size_t flat = gid * 4;
    size_t nrow = flat >> 9;             // / 512
    int    k0   = (int)(flat & 511);
    int    idx  = (int)out[O_IDX + nrow];
    float* p = out + O_ENC + flat;
    p[0] = (k0     == idx) ? 1.0f : 0.0f;
    p[1] = (k0 + 1 == idx) ? 1.0f : 0.0f;
    p[2] = (k0 + 2 == idx) ? 1.0f : 0.0f;
    p[3] = (k0 + 3 == idx) ? 1.0f : 0.0f;
}

// ---------------------------------------------------------------------------
// K3: finalize. Deterministic tree reductions: loss partials -> means,
// histogram -> perplexity. One WG of 512 threads.
// ---------------------------------------------------------------------------
__global__ void __launch_bounds__(512) vq_fin(const float* __restrict__ lossp,
                                              const unsigned* __restrict__ cnt,
                                              float* __restrict__ out) {
    __shared__ float red[512];
    const int tid = threadIdx.x;

    red[tid] = lossp[tid] + lossp[tid + 512];
    __syncthreads();
    for (int s = 256; s > 0; s >>= 1) {
        if (tid < s) red[tid] += red[tid + s];
        __syncthreads();
    }
    if (tid == 0) {
        float m = red[0] / (float)QELEMS;    // mean((q - x)^2)
        out[O_EL] = m;                       // embedding_loss
        out[O_CL] = 0.25f * m;               // commitment_loss
    }
    __syncthreads();

    float p = (float)cnt[tid] / (float)NVEC;
    red[tid] = p * logf(p + 1e-10f);
    __syncthreads();
    for (int s = 256; s > 0; s >>= 1) {
        if (tid < s) red[tid] += red[tid + s];
        __syncthreads();
    }
    if (tid == 0) out[O_P] = expf(-red[0]);
}

// ---------------------------------------------------------------------------
extern "C" void kernel_launch(void* const* d_in, const int* in_sizes, int n_in,
                              void* d_out, int out_size, void* d_ws, size_t ws_size,
                              hipStream_t stream) {
    (void)in_sizes; (void)n_in; (void)out_size; (void)ws_size;
    const float* x  = (const float*)d_in[0];   // inputs  [16, 64, 8192]
    const float* cb = (const float*)d_in[1];   // codebook [512, 64]
    float* out = (float*)d_out;

    char* ws = (char*)d_ws;
    unsigned* wsA   = (unsigned*)(ws + WS_A);
    float*    e2    = (float*)(ws + WS_E2);
    unsigned* cnt   = (unsigned*)(ws + WS_CNT);
    float*    lossp = (float*)(ws + WS_LP);

    vq_init<<<64, 256, 0, stream>>>(cb, wsA, e2, cnt);
    vq_main<<<NVEC / 128, 256, 0, stream>>>(x, cb, wsA, e2, cnt, lossp, out);
    vq_enc<<<((size_t)NVEC * NUM_EMB) / 1024, 256, 0, stream>>>(out);
    vq_fin<<<1, 512, 0, stream>>>(lossp, cnt, out);
}
